// LayerLSTM_66941360275694
// MI455X (gfx1250) — matrix operand order repaired and checked by reading, not compiled, not run on hardware
//
#include <hip/hip_runtime.h>
#include <hip/hip_bf16.h>
#include <math.h>

// ---------------------------------------------------------------------------
// 2-layer LSTM, T=512, B=64, I=H=768, per CDNA5 (gfx1250) WMMA bf16 path.
// ---------------------------------------------------------------------------

#define T_STEPS 512
#define BATCH   64
#define HID     768
#define LAYERS  2
#define GATES4  (4 * HID)          // 3072
#define BH      (BATCH * HID)      // 49152
#define KT_CNT  (HID / 32)         // 24 k-tiles of 32
#define NT_CNT  (GATES4 / 16)      // 192 gate column tiles
#define FRAG_E  512                // 32 lanes * 16 bf16 per B fragment
#define MAT_E   ((size_t)NT_CNT * KT_CNT * FRAG_E)  // 2,359,296 bf16 / matrix
#define XSEQ_E  ((size_t)T_STEPS * BATCH * HID)     // 25,165,824 elements

typedef __attribute__((ext_vector_type(16))) __bf16 v16bf;
typedef __attribute__((ext_vector_type(8)))  float  v8f;

struct U32x8 { uint4 a, b; };            // 32 bytes == v16bf
struct BF16x4 { __bf16 x, y, z, w; };    // 8 bytes

// Native converter (lowers to v_cvt_*bf16* on gfx1250, RNE)
__device__ __forceinline__ __bf16 f2bf(float f) { return (__bf16)f; }

// B fragment: 16 contiguous bf16 (pre-packed in WMMA B layout)
__device__ __forceinline__ v16bf load_b_frag(const __bf16* p) {
  U32x8 t;
  t.a = *(const uint4*)(p);
  t.b = *(const uint4*)(p + 8);
  return __builtin_bit_cast(v16bf, t);
}

// A fragment from bf16 row-major activations: lane holds K=[kb..kb+7],[kb+16..kb+23]
__device__ __forceinline__ v16bf load_a_bf16(const __bf16* row, int kbase) {
  U32x8 t;
  t.a = *(const uint4*)(row + kbase);
  t.b = *(const uint4*)(row + kbase + 16);
  return __builtin_bit_cast(v16bf, t);
}

// A fragment from f32 row-major activations (hardware cvt in registers)
__device__ __forceinline__ v16bf load_a_f32(const float* row, int kbase) {
  float4 f0 = *(const float4*)(row + kbase);
  float4 f1 = *(const float4*)(row + kbase + 4);
  float4 f2 = *(const float4*)(row + kbase + 16);
  float4 f3 = *(const float4*)(row + kbase + 20);
  v16bf r;
  r[0]  = f2bf(f0.x); r[1]  = f2bf(f0.y); r[2]  = f2bf(f0.z); r[3]  = f2bf(f0.w);
  r[4]  = f2bf(f1.x); r[5]  = f2bf(f1.y); r[6]  = f2bf(f1.z); r[7]  = f2bf(f1.w);
  r[8]  = f2bf(f2.x); r[9]  = f2bf(f2.y); r[10] = f2bf(f2.z); r[11] = f2bf(f2.w);
  r[12] = f2bf(f3.x); r[13] = f2bf(f3.y); r[14] = f2bf(f3.z); r[15] = f2bf(f3.w);
  return r;
}

__device__ __forceinline__ float sigmf(float x) {
  return 1.0f / (1.0f + __expf(-x));
}

// ---------------------------------------------------------------------------
// One-time: pack W_ih / W_hh (f32, [L][4H][K] row-major) into bf16 WMMA
// B-layout fragments: packed[(((mi*NT + nt)*KT + kt)*32 + lane)*16 + e]
//   = W[mi][ nt*16 + lane%16 ][ kt*32 + (lane/16)*8 + (e%8) + (e/8)*16 ]
// mi = layer*2 + {0:ih, 1:hh}
// ---------------------------------------------------------------------------
__global__ __launch_bounds__(256) void pack_weights(
    const float* __restrict__ W_ih, const float* __restrict__ W_hh,
    __bf16* __restrict__ wpack)
{
  int idx = blockIdx.x * 256 + threadIdx.x;       // 0 .. 4*MAT_E-1
  int e    = idx & 15;
  int t    = idx >> 4;
  int lane = t & 31;   t >>= 5;
  int kt   = t % KT_CNT; t /= KT_CNT;
  int nt   = t % NT_CNT; t /= NT_CNT;
  int mi   = t;                                   // 0..3
  int l    = mi >> 1;
  int n    = nt * 16 + (lane & 15);
  int k    = kt * 32 + ((lane >> 4) << 3) + (e & 7) + ((e >> 3) << 4);
  const float* src = (mi & 1) ? W_hh : W_ih;
  float v = src[((size_t)l * GATES4 + n) * HID + k];
  wpack[idx] = f2bf(v);
}

// ---------------------------------------------------------------------------
// One-time: convert the whole event_seq [T][B][I] f32 -> bf16 (row-major).
// Hoists all layer-0 A conversions out of the 512-step recurrence.
// ---------------------------------------------------------------------------
__global__ __launch_bounds__(256) void convert_xseq(
    const float* __restrict__ x, __bf16* __restrict__ xbf)
{
  size_t idx = (size_t)blockIdx.x * 1024 + (size_t)threadIdx.x * 4; // 4 elems/thread
  float4 f = *(const float4*)(x + idx);
  BF16x4 p;
  p.x = f2bf(f.x); p.y = f2bf(f.y); p.z = f2bf(f.z); p.w = f2bf(f.w);
  *(BF16x4*)(xbf + idx) = p;
}

// ---------------------------------------------------------------------------
// One-time: seed running states. h_state/c_state live in d_out's h_f/c_f
// regions; hbf holds bf16 h double-buffers [L][2][B*H], parity 0 seeded.
// ---------------------------------------------------------------------------
__global__ __launch_bounds__(256) void init_state(
    const float* __restrict__ h0, const float* __restrict__ c0,
    float* __restrict__ h_state, float* __restrict__ c_state,
    __bf16* __restrict__ hbf)
{
  int idx = blockIdx.x * 256 + threadIdx.x;       // 0 .. L*B*H-1
  if (idx >= LAYERS * BH) return;
  float hv = h0[idx];
  h_state[idx] = hv;
  c_state[idx] = c0[idx];
  int l = idx / BH;
  int r = idx - l * BH;
  hbf[((size_t)l * 2 + 0) * BH + r] = f2bf(hv);
}

// ---------------------------------------------------------------------------
// One layer, one timestep, fused: gates GEMM (WMMA bf16) + LSTM cell update.
// Grid: 48 blocks (one per 16 hidden columns), 256 threads = 8 waves.
// Wave w: m_tile = w&3, gates {2*(w>>2), 2*(w>>2)+1}; full K=768 twice.
// ---------------------------------------------------------------------------
template <bool X_F32>
__global__ __launch_bounds__(256, 1) void lstm_layer_step(
    const float*  __restrict__ xf32,    // X_F32 path: event_seq + t*B*I
    const __bf16* __restrict__ xbf,     // bf16 path: pre-converted x rows
    const __bf16* __restrict__ hbf_in,  // h_{t-1} of this layer (bf16)
    const __bf16* __restrict__ wih,     // packed bf16 W_ih
    const __bf16* __restrict__ whh,     // packed bf16 W_hh
    const float*  __restrict__ bih,
    const float*  __restrict__ bhh,
    const int*    __restrict__ length,
    float*        __restrict__ h_state, // f32 running h (d_out h_f slice)
    float*        __restrict__ c_state, // f32 running c (d_out c_f slice)
    __bf16*       __restrict__ hbf_out, // h_t bf16 (other parity buffer)
    float*        __restrict__ out,     // d_out base, [T][B][L][H]
    int t, int l)
{
  __shared__ float gates_lds[64 * 65];  // [gate*16+ncol][m], padded stride

  const int tid    = threadIdx.x;
  const int lane   = tid & 31;
  const int w      = tid >> 5;
  const int m_tile = w & 3;
  const int g0     = (w >> 2) << 1;     // 0 or 2
  const int g1     = g0 + 1;
  const int j      = blockIdx.x;        // hidden-column tile 0..47
  const int m      = m_tile * 16 + (lane & 15);
  const int koff   = (lane >> 4) << 3;  // A-frag per-lane K base
  const int nt0    = g0 * 48 + j;       // gate-column tiles (each gate = 48 tiles)
  const int nt1    = g1 * 48 + j;

  v8f acc0 = {};
  v8f acc1 = {};

  // ---- phase 1: x @ W_ih^T ----
  {
    const __bf16* w0 = wih + (size_t)nt0 * (KT_CNT * FRAG_E) + lane * 16;
    const __bf16* w1 = wih + (size_t)nt1 * (KT_CNT * FRAG_E) + lane * 16;
    if (X_F32) {
      const float* xrow = xf32 + (size_t)m * HID;
#pragma unroll 4
      for (int kt = 0; kt < KT_CNT; ++kt) {
        v16bf a  = load_a_f32(xrow, kt * 32 + koff);
        v16bf b0 = load_b_frag(w0 + kt * FRAG_E);
        v16bf b1 = load_b_frag(w1 + kt * FRAG_E);
        acc0 = __builtin_amdgcn_wmma_f32_16x16x32_bf16(false, a, false, b0, (short)0, acc0, false, false);
        acc1 = __builtin_amdgcn_wmma_f32_16x16x32_bf16(false, a, false, b1, (short)0, acc1, false, false);
      }
    } else {
      const __bf16* xrow = xbf + (size_t)m * HID;
#pragma unroll 4
      for (int kt = 0; kt < KT_CNT; ++kt) {
        v16bf a  = load_a_bf16(xrow, kt * 32 + koff);
        v16bf b0 = load_b_frag(w0 + kt * FRAG_E);
        v16bf b1 = load_b_frag(w1 + kt * FRAG_E);
        acc0 = __builtin_amdgcn_wmma_f32_16x16x32_bf16(false, a, false, b0, (short)0, acc0, false, false);
        acc1 = __builtin_amdgcn_wmma_f32_16x16x32_bf16(false, a, false, b1, (short)0, acc1, false, false);
      }
    }
  }

  // ---- phase 2: h_{t-1} @ W_hh^T ----
  {
    const __bf16* hrow = hbf_in + (size_t)m * HID;
    const __bf16* v0 = whh + (size_t)nt0 * (KT_CNT * FRAG_E) + lane * 16;
    const __bf16* v1 = whh + (size_t)nt1 * (KT_CNT * FRAG_E) + lane * 16;
#pragma unroll 4
    for (int kt = 0; kt < KT_CNT; ++kt) {
      v16bf a  = load_a_bf16(hrow, kt * 32 + koff);
      v16bf b0 = load_b_frag(v0 + kt * FRAG_E);
      v16bf b1 = load_b_frag(v1 + kt * FRAG_E);
      acc0 = __builtin_amdgcn_wmma_f32_16x16x32_bf16(false, a, false, b0, (short)0, acc0, false, false);
      acc1 = __builtin_amdgcn_wmma_f32_16x16x32_bf16(false, a, false, b1, (short)0, acc1, false, false);
    }
  }

  // ---- accumulators -> LDS (C layout: VGPR r, lane L -> M=r+8*(L/16), N=L%16)
  {
    const int nlocal = lane & 15;
    const int mbase  = m_tile * 16 + ((lane >> 4) << 3);
    float* p0 = &gates_lds[(g0 * 16 + nlocal) * 65 + mbase];
    float* p1 = &gates_lds[(g1 * 16 + nlocal) * 65 + mbase];
#pragma unroll
    for (int r = 0; r < 8; ++r) { p0[r] = acc0[r]; p1[r] = acc1[r]; }
  }
  __syncthreads();

  // ---- fused LSTM cell update: 16 hidden cols x 64 batches = 1024 elements
#pragma unroll
  for (int it = 0; it < 4; ++it) {
    int e   = tid + it * 256;
    int b   = e >> 4;
    int col = e & 15;
    int hh  = j * 16 + col;

    float ig = gates_lds[( 0 + col) * 65 + b] + bih[           hh] + bhh[           hh];
    float fg = gates_lds[(16 + col) * 65 + b] + bih[    HID + hh] + bhh[    HID + hh];
    float gg = gates_lds[(32 + col) * 65 + b] + bih[2 * HID + hh] + bhh[2 * HID + hh];
    float og = gates_lds[(48 + col) * 65 + b] + bih[3 * HID + hh] + bhh[3 * HID + hh];

    int   si    = b * HID + hh;
    float c_old = c_state[si];
    float h_old = h_state[si];

    float c_new = sigmf(fg) * c_old + sigmf(ig) * tanhf(gg);
    float h_new = sigmf(og) * tanhf(c_new);

    float msk = (t < length[b]) ? 1.0f : 0.0f;
    h_new = msk * h_new + (1.0f - msk) * h_old;
    c_new = msk * c_new + (1.0f - msk) * c_old;

    c_state[si] = c_new;
    h_state[si] = h_new;
    hbf_out[si] = f2bf(h_new);
    out[(((size_t)t * BATCH + b) * LAYERS + l) * HID + hh] = h_new;
  }
}

// ---------------------------------------------------------------------------
// Host side
// ---------------------------------------------------------------------------
extern "C" void kernel_launch(void* const* d_in, const int* in_sizes, int n_in,
                              void* d_out, int out_size, void* d_ws, size_t ws_size,
                              hipStream_t stream) {
  const float* event_seq = (const float*)d_in[0];   // [T][B][I]
  const float* h0        = (const float*)d_in[1];   // [L][B][H]
  const float* c0        = (const float*)d_in[2];   // [L][B][H]
  const float* W_ih      = (const float*)d_in[3];   // [L][4H][I]
  const float* W_hh      = (const float*)d_in[4];   // [L][4H][H]
  const float* b_ih      = (const float*)d_in[5];   // [L][4H]
  const float* b_hh      = (const float*)d_in[6];   // [L][4H]
  const int*   length    = (const int*)  d_in[7];   // [B]

  float* out     = (float*)d_out;                       // [T][B][L][H]
  float* h_state = out + (size_t)T_STEPS * BATCH * LAYERS * HID; // h_f [L][B][H]
  float* c_state = h_state + (size_t)LAYERS * BH;               // c_f [L][B][H]

  // Workspace layout (bf16 elements):
  //   [0)            4*MAT_E   packed weights          (18.9 MB)
  //   [4*MAT_E)      4*BH      h double buffers        (384 KB)
  //   [.. optional)  XSEQ_E    pre-converted event_seq (50.3 MB)
  __bf16* wpack = (__bf16*)d_ws;
  __bf16* hbf   = wpack + 4 * MAT_E;   // [L][2 parity][B*H]
  __bf16* xbf   = hbf + 4 * (size_t)BH;

  const size_t ws_need_full = (4 * MAT_E + 4 * (size_t)BH + XSEQ_E) * sizeof(__bf16);
  const bool   use_xbf      = (ws_size >= ws_need_full);

  // one-time per call (deterministic): pack weights, seed states, convert x
  pack_weights<<<dim3((unsigned)(4 * MAT_E / 256)), dim3(256), 0, stream>>>(
      W_ih, W_hh, wpack);
  init_state<<<dim3((LAYERS * BH + 255) / 256), dim3(256), 0, stream>>>(
      h0, c0, h_state, c_state, hbf);
  if (use_xbf) {
    convert_xseq<<<dim3((unsigned)(XSEQ_E / 1024)), dim3(256), 0, stream>>>(
        event_seq, xbf);
  }

  for (int t = 0; t < T_STEPS; ++t) {
    const int rp = t & 1;        // read parity
    const int wp = rp ^ 1;       // write parity

    // layer 0: x = event_seq[t] (pre-converted bf16 if workspace allows)
    if (use_xbf) {
      lstm_layer_step<false><<<dim3(48), dim3(256), 0, stream>>>(
          (const float*)nullptr, xbf + (size_t)t * BH,
          hbf + ((size_t)0 * 2 + rp) * BH,
          wpack + 0 * MAT_E, wpack + 1 * MAT_E,
          b_ih + 0, b_hh + 0, length,
          h_state + 0, c_state + 0,
          hbf + ((size_t)0 * 2 + wp) * BH,
          out, t, 0);
    } else {
      lstm_layer_step<true><<<dim3(48), dim3(256), 0, stream>>>(
          event_seq + (size_t)t * BH, (const __bf16*)nullptr,
          hbf + ((size_t)0 * 2 + rp) * BH,
          wpack + 0 * MAT_E, wpack + 1 * MAT_E,
          b_ih + 0, b_hh + 0, length,
          h_state + 0, c_state + 0,
          hbf + ((size_t)0 * 2 + wp) * BH,
          out, t, 0);
    }

    // layer 1: x = layer0 h_t (bf16, just written)
    lstm_layer_step<false><<<dim3(48), dim3(256), 0, stream>>>(
        (const float*)nullptr, hbf + ((size_t)0 * 2 + wp) * BH,
        hbf + ((size_t)1 * 2 + rp) * BH,
        wpack + 2 * MAT_E, wpack + 3 * MAT_E,
        b_ih + GATES4, b_hh + GATES4, length,
        h_state + BH, c_state + BH,
        hbf + ((size_t)1 * 2 + wp) * BH,
        out, t, 1);
  }
}
